// GraphModel_23192823399051
// MI455X (gfx1250) — matrix operand order repaired and checked by reading, not compiled, not run on hardware
//
#include <hip/hip_runtime.h>
#include <hip/hip_bf16.h>
#include <math.h>

typedef _Float16 h16;
typedef __attribute__((ext_vector_type(8)))  _Float16 v8h;
typedef __attribute__((ext_vector_type(16))) _Float16 v16h;
typedef __attribute__((ext_vector_type(8)))  float    v8f;

// ---------------------------------------------------------------------------
// WMMA fragment helpers (CDNA5 wave32 layouts, per CDNA5 ISA)
// A (16x32 f16): lanes 0-15 -> row M=lane, K halves {c..c+7, c+16..c+23};
//                lanes 16-31 -> row M=lane-16, K halves {c+8..c+15, c+24..c+31}
// B (32x16 f16): lane holds column N=col0+(lane&15), K in same order as A
// C/D (16x16 f32): elem r of v8f -> (row = r + 8*(lane>=16), col = lane&15)
// ---------------------------------------------------------------------------
__device__ __forceinline__ v16h load_a_frag(const h16* __restrict__ base, int stride,
                                            int row0, int c0) {
  int lane = threadIdx.x & 31;
  const h16* p = base + (size_t)(row0 + (lane & 15)) * stride + (c0 + ((lane >> 4) << 3));
  union { v16h v; v8h h[2]; } u;
  u.h[0] = *(const v8h*)(p);
  u.h[1] = *(const v8h*)(p + 16);
  return u.v;
}

__device__ __forceinline__ v16h load_b_frag(const h16* __restrict__ base, int stride,
                                            int c0, int col0) {
  int lane = threadIdx.x & 31;
  const h16* p = base + (size_t)(c0 + ((lane >> 4) << 3)) * stride + col0 + (lane & 15);
  v16h v;
#pragma unroll
  for (int s = 0; s < 8; s++) {
    v[s] = p[(size_t)s * stride];
    v[8 + s] = p[(size_t)(16 + s) * stride];
  }
  return v;
}

// error-compensated f16 split matmul: acc += Ah*Bh + Ah*Bl + Al*Bh  (~f32 accuracy)
__device__ __forceinline__ v8f wmma_split(v16h ah, v16h al, v16h bh, v16h bl, v8f acc) {
  acc = __builtin_amdgcn_wmma_f32_16x16x32_f16(false, ah, false, bh, (short)0, acc, false, false);
  acc = __builtin_amdgcn_wmma_f32_16x16x32_f16(false, ah, false, bl, (short)0, acc, false, false);
  acc = __builtin_amdgcn_wmma_f32_16x16x32_f16(false, al, false, bh, (short)0, acc, false, false);
  return acc;
}

// ---------------------------------------------------------------------------
// Conversion kernels
// ---------------------------------------------------------------------------
// f32 (B,N,C) [src row stride may differ, e.g. a slice of cat] -> split f16
// packed (B,N,C) and optionally transposed (B,C,N)
__global__ __launch_bounds__(256) void k_convert(const float* __restrict__ src, int srcStride,
                                                 int C, int N, long total,
                                                 h16* __restrict__ Fh, h16* __restrict__ Fl,
                                                 h16* __restrict__ FhT, h16* __restrict__ FlT,
                                                 int doT) {
  long i = (long)blockIdx.x * 256 + threadIdx.x;
  if (i >= total) return;
  int c = (int)(i % C);
  long bn = i / C;
  int n = (int)(bn % N);
  int b = (int)(bn / N);
  float x = src[bn * srcStride + c];
  h16 hi = (h16)x;
  h16 lo = (h16)(x - (float)hi);
  Fh[i] = hi;
  Fl[i] = lo;
  if (doT) {
    long ti = ((long)b * C + c) * N + n;
    FhT[ti] = hi;
    FlT[ti] = lo;
  }
}

// per-point squared norm (f32, exact)
__global__ __launch_bounds__(256) void k_xx(const float* __restrict__ src, int srcStride,
                                            int C, float* __restrict__ xx) {
  int i = blockIdx.x * 256 + threadIdx.x; // b*N + n
  const float* p = src + (size_t)i * srcStride;
  float s = 0.f;
  for (int c = 0; c < C; c++) { float v = p[c]; s = fmaf(v, v, s); }
  xx[i] = s;
}

// weights (O, dual?2C:C) -> transposed split f16 (C,O); delta = w2-w1 when dual
__global__ __launch_bounds__(256) void k_wconv(const float* __restrict__ w, int O, int Cin,
                                               int dual,
                                               h16* __restrict__ W1h, h16* __restrict__ W1l,
                                               h16* __restrict__ Wdh, h16* __restrict__ Wdl) {
  int i = blockIdx.x * 256 + threadIdx.x; // c*O + o
  if (i >= Cin * O) return;
  int c = i / O, o = i % O;
  int ws = dual ? (2 * Cin) : Cin;
  float w1 = w[(size_t)o * ws + c];
  h16 h1 = (h16)w1;
  W1h[i] = h1;
  W1l[i] = (h16)(w1 - (float)h1);
  if (dual) {
    float wd = w[(size_t)o * ws + Cin + c] - w1;
    h16 hd = (h16)wd;
    Wdh[i] = hd;
    Wdl[i] = (h16)(wd - (float)hd);
  }
}

// ---------------------------------------------------------------------------
// Fused KNN: WMMA inner products (split-f16) + per-row top-20 (sorted regs)
// Block = 256 threads (8 waves) = 256 rows; column tiles of 32 through LDS.
// ---------------------------------------------------------------------------
__global__ __launch_bounds__(256) void k_knn(const h16* __restrict__ Fh, const h16* __restrict__ Fl,
                                             const h16* __restrict__ FhT, const h16* __restrict__ FlT,
                                             const float* __restrict__ xx,
                                             int C, int N, int* __restrict__ idxout) {
  __shared__ float pdbuf[256][33];
  __shared__ float xxrow[256];
  int b = blockIdx.y;
  int row0 = blockIdx.x * 256;
  int t = threadIdx.x;
  int wave = t >> 5, lane = t & 31;
  int colL = lane & 15, rh = (lane >> 4) << 3;
  int wrow = row0 + wave * 32;           // wave's 32-row base (global)
  int myrow = row0 + t;                  // this thread's row for the scan phase
  const h16* fh  = Fh  + (size_t)b * N * C;
  const h16* fl  = Fl  + (size_t)b * N * C;
  const h16* fht = FhT + (size_t)b * C * N;
  const h16* flt = FlT + (size_t)b * C * N;
  const float* xb = xx + (size_t)b * N;

  xxrow[t] = xb[row0 + t];

  float kv[20];
  int   ki[20];
#pragma unroll
  for (int q = 0; q < 20; q++) { kv[q] = -INFINITY; ki[q] = 0; }

  __syncthreads();

  for (int ct = 0; ct < N; ct += 32) {
    v8f a00 = {}, a01 = {}, a10 = {}, a11 = {};
    for (int c0 = 0; c0 < C; c0 += 32) {
      v16h A0h = load_a_frag(fh, C, wrow, c0);
      v16h A0l = load_a_frag(fl, C, wrow, c0);
      v16h A1h = load_a_frag(fh, C, wrow + 16, c0);
      v16h A1l = load_a_frag(fl, C, wrow + 16, c0);
      v16h B0h = load_b_frag(fht, N, c0, ct);
      v16h B0l = load_b_frag(flt, N, c0, ct);
      v16h B1h = load_b_frag(fht, N, c0, ct + 16);
      v16h B1l = load_b_frag(flt, N, c0, ct + 16);
      a00 = wmma_split(A0h, A0l, B0h, B0l, a00);
      a01 = wmma_split(A0h, A0l, B1h, B1l, a01);
      a10 = wmma_split(A1h, A1l, B0h, B0l, a10);
      a11 = wmma_split(A1h, A1l, B1h, B1l, a11);
    }
    // pd = 2*inner - |x_n|^2 - |x_m|^2  -> LDS
#pragma unroll
    for (int r = 0; r < 8; r++) {
      int lr0 = wave * 32 + rh + r;        // rows of subtile pair 0
      int lr1 = lr0 + 16;                  // rows of subtile pair 1
      float xc0 = xb[ct + colL];
      float xc1 = xb[ct + 16 + colL];
      pdbuf[lr0][colL]      = 2.f * a00[r] - xxrow[lr0] - xc0;
      pdbuf[lr0][16 + colL] = 2.f * a01[r] - xxrow[lr0] - xc1;
      pdbuf[lr1][colL]      = 2.f * a10[r] - xxrow[lr1] - xc0;
      pdbuf[lr1][16 + colL] = 2.f * a11[r] - xxrow[lr1] - xc1;
    }
    __syncthreads();
    // each thread scans 32 candidates for its own row
    for (int j = 0; j < 32; j++) {
      int gcol = ct + j;
      float v = pdbuf[t][j];
      if (gcol == myrow) v = 0.0f;         // exact diagonal (self), max of row
      if (v > kv[19]) {                    // sorted-descending bubble insert
        float cv = v; int ci = gcol;
#pragma unroll
        for (int q = 0; q < 20; q++) {
          if (cv > kv[q]) {
            float tv = kv[q]; int ti = ki[q];
            kv[q] = cv; ki[q] = ci;
            cv = tv; ci = ti;
          }
        }
      }
    }
    __syncthreads();
  }
  int* op = idxout + ((size_t)b * N + myrow) * 20;
#pragma unroll
  for (int q = 0; q < 20; q++) op[q] = ki[q];
}

// ---------------------------------------------------------------------------
// g = F * W1^T,  t = F * (W2-W1)^T   (split-f16 WMMA, f32 out)
// ---------------------------------------------------------------------------
__global__ __launch_bounds__(256) void k_gt(const h16* __restrict__ Fh, const h16* __restrict__ Fl,
                                            int C,
                                            const h16* __restrict__ W1h, const h16* __restrict__ W1l,
                                            const h16* __restrict__ Wdh, const h16* __restrict__ Wdl,
                                            int O, int N,
                                            float* __restrict__ g, float* __restrict__ tt) {
  int b = blockIdx.y;
  int wave = threadIdx.x >> 5, lane = threadIdx.x & 31;
  int colL = lane & 15, rh = (lane >> 4) << 3;
  int r0 = blockIdx.x * 256 + wave * 32;
  const h16* fh = Fh + (size_t)b * N * C;
  const h16* fl = Fl + (size_t)b * N * C;
  float* gp = g  + (size_t)b * N * O;
  float* tp = tt + (size_t)b * N * O;

  for (int o0 = 0; o0 < O; o0 += 16) {
    v8f ag0 = {}, ag1 = {}, at0 = {}, at1 = {};
    for (int c0 = 0; c0 < C; c0 += 32) {
      v16h A0h = load_a_frag(fh, C, r0, c0);
      v16h A0l = load_a_frag(fl, C, r0, c0);
      v16h A1h = load_a_frag(fh, C, r0 + 16, c0);
      v16h A1l = load_a_frag(fl, C, r0 + 16, c0);
      v16h B1h = load_b_frag(W1h, O, c0, o0);
      v16h B1l = load_b_frag(W1l, O, c0, o0);
      v16h BDh = load_b_frag(Wdh, O, c0, o0);
      v16h BDl = load_b_frag(Wdl, O, c0, o0);
      ag0 = wmma_split(A0h, A0l, B1h, B1l, ag0);
      ag1 = wmma_split(A1h, A1l, B1h, B1l, ag1);
      at0 = wmma_split(A0h, A0l, BDh, BDl, at0);
      at1 = wmma_split(A1h, A1l, BDh, BDl, at1);
    }
#pragma unroll
    for (int r = 0; r < 8; r++) {
      size_t ro0 = (size_t)(r0 + rh + r) * O + o0 + colL;
      size_t ro1 = (size_t)(r0 + rh + r + 16) * O + o0 + colL;
      gp[ro0] = ag0[r]; gp[ro1] = ag1[r];
      tp[ro0] = at0[r]; tp[ro1] = at1[r];
    }
  }
}

// ---------------------------------------------------------------------------
// out_cat[b,n,cOff+o] = max_k leaky(bn(g[idx[n,k],o] + t[n,o]))
// one block per point, blockDim = O (g table is L2-resident)
// ---------------------------------------------------------------------------
__global__ __launch_bounds__(256) void k_gather(const float* __restrict__ g,
                                                const float* __restrict__ tt,
                                                const int* __restrict__ idx,
                                                int O, int N,
                                                const float* __restrict__ bg, const float* __restrict__ bbv,
                                                const float* __restrict__ bm, const float* __restrict__ bv,
                                                float* __restrict__ cat, int cOff) {
  __shared__ int nb[20];
  int bn = blockIdx.x;      // b*N + n
  int b = bn / N;
  int o = threadIdx.x;
  if (threadIdx.x < 20) nb[threadIdx.x] = idx[(size_t)bn * 20 + threadIdx.x];
  __syncthreads();
  float s = bg[o] * rsqrtf(bv[o] + 1e-5f);
  float bias = bbv[o] - bm[o] * s;
  const float* gb = g + (size_t)b * N * O;
  float tv = tt[(size_t)bn * O + o];
  float mx = -INFINITY;
#pragma unroll 5
  for (int k = 0; k < 20; k++) {
    float y = gb[(size_t)nb[k] * O + o] + tv;
    float z = y * s + bias;
    z = (z >= 0.f) ? z : 0.2f * z;
    mx = fmaxf(mx, z);
  }
  cat[(size_t)bn * 512 + cOff + o] = mx;
}

// ---------------------------------------------------------------------------
// x5 = leaky(bn(cat @ w5^T)) fused with max/mean pooling over n (per n-chunk
// partials; deterministic reduce afterwards). 256 rows x 64 cols per block.
// ---------------------------------------------------------------------------
__global__ __launch_bounds__(256) void k_pool(const h16* __restrict__ CAh, const h16* __restrict__ CAl,
                                              const h16* __restrict__ Wh, const h16* __restrict__ Wl,
                                              const float* __restrict__ bg, const float* __restrict__ bbv,
                                              const float* __restrict__ bm, const float* __restrict__ bv,
                                              float* __restrict__ pmaxp, float* __restrict__ psump) {
  const int N = 4096, C = 512, O = 1024;
  __shared__ float redmax[8][4][16];
  __shared__ float redsum[8][4][16];
  int b = blockIdx.z;
  int ci = blockIdx.x;                 // n-chunk index (0..15)
  int o0 = blockIdx.y * 64;
  int wave = threadIdx.x >> 5, lane = threadIdx.x & 31;
  int colL = lane & 15;
  int r0 = ci * 256 + wave * 32;
  const h16* fh = CAh + (size_t)b * N * C;
  const h16* fl = CAl + (size_t)b * N * C;

  v8f acc[4][2];
#pragma unroll
  for (int cs = 0; cs < 4; cs++) { acc[cs][0] = {}; acc[cs][1] = {}; }

  for (int c0 = 0; c0 < C; c0 += 32) {
    v16h A0h = load_a_frag(fh, C, r0, c0);
    v16h A0l = load_a_frag(fl, C, r0, c0);
    v16h A1h = load_a_frag(fh, C, r0 + 16, c0);
    v16h A1l = load_a_frag(fl, C, r0 + 16, c0);
#pragma unroll
    for (int cs = 0; cs < 4; cs++) {
      v16h Bh = load_b_frag(Wh, O, c0, o0 + cs * 16);
      v16h Bl = load_b_frag(Wl, O, c0, o0 + cs * 16);
      acc[cs][0] = wmma_split(A0h, A0l, Bh, Bl, acc[cs][0]);
      acc[cs][1] = wmma_split(A1h, A1l, Bh, Bl, acc[cs][1]);
    }
  }
#pragma unroll
  for (int cs = 0; cs < 4; cs++) {
    int col = o0 + cs * 16 + colL;
    float s = bg[col] * rsqrtf(bv[col] + 1e-5f);
    float bias = bbv[col] - bm[col] * s;
    float lmax = -INFINITY, lsum = 0.f;
#pragma unroll
    for (int rs = 0; rs < 2; rs++)
#pragma unroll
      for (int r = 0; r < 8; r++) {
        float z = acc[cs][rs][r] * s + bias;
        z = (z >= 0.f) ? z : 0.2f * z;
        lmax = fmaxf(lmax, z);
        lsum += z;
      }
    lmax = fmaxf(lmax, __shfl_xor(lmax, 16, 32));
    lsum = lsum + __shfl_xor(lsum, 16, 32);
    if (lane < 16) { redmax[wave][cs][lane] = lmax; redsum[wave][cs][lane] = lsum; }
  }
  __syncthreads();
  if (threadIdx.x < 64) {
    int cs = threadIdx.x >> 4, cl = threadIdx.x & 15;
    float m = redmax[0][cs][cl], sm = redsum[0][cs][cl];
#pragma unroll
    for (int w = 1; w < 8; w++) { m = fmaxf(m, redmax[w][cs][cl]); sm += redsum[w][cs][cl]; }
    size_t pi = ((size_t)b * 16 + ci) * 1024 + o0 + cs * 16 + cl;
    pmaxp[pi] = m;
    psump[pi] = sm;
  }
}

__global__ __launch_bounds__(256) void k_poolred(const float* __restrict__ pmaxp,
                                                 const float* __restrict__ psump,
                                                 float* __restrict__ pmax, float* __restrict__ psum) {
  int i = blockIdx.x * 256 + threadIdx.x; // b*1024 + o
  int b = i >> 10, o = i & 1023;
  float m = -INFINITY, s = 0.f;
#pragma unroll
  for (int c = 0; c < 16; c++) {
    size_t pi = ((size_t)b * 16 + c) * 1024 + o;
    m = fmaxf(m, pmaxp[pi]);
    s += psump[pi];
  }
  pmax[i] = m;
  psum[i] = s;
}

// ---------------------------------------------------------------------------
// final FC + bn + leaky (tiny: 33 MFLOP, plain f32)
// ---------------------------------------------------------------------------
__global__ __launch_bounds__(256) void k_fc(const float* __restrict__ pmax, const float* __restrict__ psum,
                                            const float* __restrict__ fcw, const float* __restrict__ fcb,
                                            const float* __restrict__ g6, const float* __restrict__ b6,
                                            const float* __restrict__ m6, const float* __restrict__ v6,
                                            float* __restrict__ out) {
  int i = blockIdx.x * 256 + threadIdx.x; // b*1024 + j
  int b = i >> 10, j = i & 1023;
  const float* w = fcw + (size_t)j * 2048;
  const float* pm = pmax + b * 1024;
  const float* ps = psum + b * 1024;
  float acc = fcb[j];
  for (int c = 0; c < 1024; c++) acc = fmaf(pm[c], w[c], acc);
  const float invN = 1.f / 4096.f;
  for (int c = 0; c < 1024; c++) acc = fmaf(ps[c] * invN, w[1024 + c], acc);
  float s = g6[j] * rsqrtf(v6[j] + 1e-5f);
  float z = (acc - m6[j]) * s + b6[j];
  out[i] = (z >= 0.f) ? z : 0.2f * z;
}

// ---------------------------------------------------------------------------
// host orchestration
// ---------------------------------------------------------------------------
extern "C" void kernel_launch(void* const* d_in, const int* in_sizes, int n_in,
                              void* d_out, int out_size, void* d_ws, size_t ws_size,
                              hipStream_t stream) {
  (void)in_sizes; (void)n_in; (void)out_size; (void)ws_size;
  const int B = 8, N = 4096;
  const float* x = (const float*)d_in[0];
  const float *W[5], *BG[5], *BBp[5], *BM[5], *BV[5];
  for (int l = 0; l < 5; l++) {
    W[l]   = (const float*)d_in[1 + 5 * l];
    BG[l]  = (const float*)d_in[2 + 5 * l];
    BBp[l] = (const float*)d_in[3 + 5 * l];
    BM[l]  = (const float*)d_in[4 + 5 * l];
    BV[l]  = (const float*)d_in[5 + 5 * l];
  }
  const float* fcw = (const float*)d_in[26];
  const float* fcb = (const float*)d_in[27];
  const float* g6 = (const float*)d_in[28];
  const float* b6 = (const float*)d_in[29];
  const float* m6 = (const float*)d_in[30];
  const float* v6 = (const float*)d_in[31];
  float* out = (float*)d_out;

  char* wsp = (char*)d_ws;
  size_t off = 0;
  auto carve = [&](size_t bytes) -> void* {
    void* r = wsp + off;
    off += (bytes + 255) & ~(size_t)255;
    return r;
  };
  int*   idxb = (int*)carve((size_t)B * N * 20 * 4);
  float* xx   = (float*)carve((size_t)B * N * 4);
  h16*   Fh   = (h16*)carve((size_t)B * N * 512 * 2);   // also holds cat-f16 hi
  h16*   Fl   = (h16*)carve((size_t)B * N * 512 * 2);   // also holds cat-f16 lo
  h16*   FhT  = (h16*)carve((size_t)B * 128 * N * 2);
  h16*   FlT  = (h16*)carve((size_t)B * 128 * N * 2);
  h16*   W1h  = (h16*)carve((size_t)512 * 1024 * 2);    // also holds w5^T hi
  h16*   W1l  = (h16*)carve((size_t)512 * 1024 * 2);    // also holds w5^T lo
  h16*   Wdh  = (h16*)carve((size_t)128 * 256 * 2);
  h16*   Wdl  = (h16*)carve((size_t)128 * 256 * 2);
  float* g    = (float*)carve((size_t)B * N * 256 * 4);
  float* t    = (float*)carve((size_t)B * N * 256 * 4);
  float* cat  = (float*)carve((size_t)B * N * 512 * 4);
  float* pmaxp = (float*)carve((size_t)B * 16 * 1024 * 4);
  float* psump = (float*)carve((size_t)B * 16 * 1024 * 4);
  float* pmax  = (float*)carve((size_t)B * 1024 * 4);
  float* psum  = (float*)carve((size_t)B * 1024 * 4);

  const int Cin[4]  = {64, 64, 64, 128};
  const int Oc[4]   = {64, 64, 128, 256};
  const int cOff[4] = {0, 64, 128, 256};

  for (int l = 0; l < 4; l++) {
    const float* Fsrc = (l == 0) ? x : (cat + cOff[l - 1]);
    int srcStride = (l == 0) ? 64 : 512;
    long totalF = (long)B * N * Cin[l];
    k_convert<<<(unsigned)((totalF + 255) / 256), 256, 0, stream>>>(
        Fsrc, srcStride, Cin[l], N, totalF, Fh, Fl, FhT, FlT, 1);
    k_xx<<<(B * N) / 256, 256, 0, stream>>>(Fsrc, srcStride, Cin[l], xx);
    int wtot = Cin[l] * Oc[l];
    k_wconv<<<(wtot + 255) / 256, 256, 0, stream>>>(W[l], Oc[l], Cin[l], 1, W1h, W1l, Wdh, Wdl);
    dim3 kg(N / 256, B);
    k_knn<<<kg, 256, 0, stream>>>(Fh, Fl, FhT, FlT, xx, Cin[l], N, idxb);
    k_gt<<<kg, 256, 0, stream>>>(Fh, Fl, Cin[l], W1h, W1l, Wdh, Wdl, Oc[l], N, g, t);
    k_gather<<<B * N, Oc[l], 0, stream>>>(g, t, idxb, Oc[l], N,
                                          BG[l], BBp[l], BM[l], BV[l], cat, cOff[l]);
  }

  long ctot = (long)B * N * 512;
  k_convert<<<(unsigned)((ctot + 255) / 256), 256, 0, stream>>>(
      cat, 512, 512, N, ctot, Fh, Fl, (h16*)nullptr, (h16*)nullptr, 0);
  k_wconv<<<(512 * 1024) / 256, 256, 0, stream>>>(W[4], 1024, 512, 0, W1h, W1l,
                                                  (h16*)nullptr, (h16*)nullptr);
  dim3 pg(N / 256, 1024 / 64, B);
  k_pool<<<pg, 256, 0, stream>>>(Fh, Fl, W1h, W1l, BG[4], BBp[4], BM[4], BV[4], pmaxp, psump);
  k_poolred<<<(B * 1024) / 256, 256, 0, stream>>>(pmaxp, psump, pmax, psum);
  k_fc<<<(B * 1024) / 256, 256, 0, stream>>>(pmax, psum, fcw, fcb, g6, b6, m6, v6, out);
}
